// BokehFluxAttnProcessor2_0_85787676770756
// MI455X (gfx1250) — matrix operand order repaired and checked by reading, not compiled, not run on hardware
//
#include <hip/hip_runtime.h>
#include <hip/hip_bf16.h>

// ---------------------------------------------------------------------------
// BokehFlux attention for MI455X (gfx1250, wave32, WMMA).
// All heavy GEMMs (QKV projections, output projections, QK^T, PV) run on
// v_wmma_f32_16x16x32_bf16: fp32 data in HBM, bf16 tiles in LDS/registers,
// fp32 accumulation.  ~275 GFLOP vs ~310MB traffic => compute bound => use
// the matrix pipe with the deepest-K 16-bit WMMA shape.
// Attention stages K/V tiles in LDS once per 4-wave block (same head),
// cutting global K/V traffic 4x and turning fragment builds into paired
// ds_load_b32 reads.
// ---------------------------------------------------------------------------

typedef __bf16 bf16;
typedef __attribute__((ext_vector_type(16))) __bf16 bf16x16;
typedef __attribute__((ext_vector_type(8)))  float  floatx8;

union Frag {
    bf16x16 v;
    bf16 e[16];
    unsigned int u[8];
};

#define WMMA_BF16(a, b, c) \
    __builtin_amdgcn_wmma_f32_16x16x32_bf16(false, (a), false, (b), (short)0, (c), false, false)

#define DM_   3072
#define H_    24
#define DH_   128
#define S_IMG 2048
#define S_TXT 512
#define S_ALL 2560   // S_TXT + S_IMG
#define R_    16

// ---------------------------------------------------------------------------
// Generic GEMM: C[M,N] = A[M,K] @ W[N,K]^T + bias[N].
// fp32 row-major everywhere; bf16 WMMA compute.
// Requires M % 128 == 0, N % 128 == 0, K % 32 == 0 (true for every use here).
// Block = 128 threads = 4 waves.  Block tile 128x128, wave tile 64x64.
// ---------------------------------------------------------------------------
__global__ __launch_bounds__(128)
void gemm_bias_kernel(const float* __restrict__ A, const float* __restrict__ W,
                      const float* __restrict__ bias, float* __restrict__ C,
                      int M, int N, int K)
{
    const int LDT = 34;                       // padded LDS row stride (bf16)
    __shared__ __align__(16) bf16 As[128 * 34];
    __shared__ __align__(16) bf16 Bs[128 * 34];

    const int tid  = threadIdx.x;
    const int lane = tid & 31;
    const int wave = tid >> 5;
    const int bm   = blockIdx.y * 128;
    const int bn   = blockIdx.x * 128;
    const int wm   = (wave >> 1) * 64;
    const int wn   = (wave & 1) * 64;
    const int ml   = lane & 15;               // row-in-tile for A / col for B
    const int kg   = lane >> 4;               // lane half

    // staging pattern: lanes 0..7 cover one contiguous 128B span of a row
    const int srow = tid >> 3;                // 0..15
    const int scol = (tid & 7) * 4;           // 0..28

    floatx8 acc[4][4];
    #pragma unroll
    for (int i = 0; i < 4; ++i)
        #pragma unroll
        for (int j = 0; j < 4; ++j)
            #pragma unroll
            for (int r = 0; r < 8; ++r) acc[i][j][r] = 0.0f;

    for (int k0 = 0; k0 < K; k0 += 32) {
        // Stage one 128x32 tile of A and of W into LDS (fp32 -> bf16).
        #pragma unroll
        for (int i = 0; i < 8; ++i) {
            const int row = srow + 16 * i;
            float4 fa = *(const float4*)(A + (size_t)(bm + row) * K + k0 + scol);
            float4 fb = *(const float4*)(W + (size_t)(bn + row) * K + k0 + scol);
            bf16* da = As + row * LDT + scol;
            bf16* db = Bs + row * LDT + scol;
            da[0] = (bf16)fa.x; da[1] = (bf16)fa.y; da[2] = (bf16)fa.z; da[3] = (bf16)fa.w;
            db[0] = (bf16)fb.x; db[1] = (bf16)fb.y; db[2] = (bf16)fb.z; db[3] = (bf16)fb.w;
        }
        __syncthreads();

        Frag a[4], b[4];
        #pragma unroll
        for (int t = 0; t < 4; ++t) {
            const bf16* ra = As + (wm + t * 16 + ml) * LDT;
            const bf16* rb = Bs + (wn + t * 16 + ml) * LDT;
            #pragma unroll
            for (int j = 0; j < 8; ++j) {
                // A layout: lanes 0-15 K {0-7,16-23}; lanes 16-31 K {8-15,24-31}
                const int kkA = ((j < 4) ? 2 * j : 16 + 2 * (j - 4)) + kg * 8;
                // B layout: n = lane&15, same K order as A
                const int kkB = ((j < 4) ? 2 * j : 16 + 2 * (j - 4)) + kg * 8;
                a[t].u[j] = *(const unsigned int*)(ra + kkA);
                b[t].u[j] = *(const unsigned int*)(rb + kkB);
            }
        }
        #pragma unroll
        for (int i = 0; i < 4; ++i)
            #pragma unroll
            for (int j = 0; j < 4; ++j)
                acc[i][j] = WMMA_BF16(a[i].v, b[j].v, acc[i][j]);
        __syncthreads();
    }

    // C/D layout: VGPR r -> row r (lanes 0-15) / r+8 (lanes 16-31), col = lane&15.
    const int hi = kg;
    #pragma unroll
    for (int i = 0; i < 4; ++i) {
        #pragma unroll
        for (int j = 0; j < 4; ++j) {
            const int n  = bn + wn + j * 16 + ml;
            const float bv = bias[n];
            #pragma unroll
            for (int r = 0; r < 8; ++r) {
                const int m = bm + wm + i * 16 + r + hi * 8;
                C[(size_t)m * N + n] = acc[i][j][r] + bv;
            }
        }
    }
}

// ---------------------------------------------------------------------------
// LoRA down:  tmp[row][r] = sum_k X[row][k] * down[r][k]   (K = 3072 fixed)
// One wave per output element.
// ---------------------------------------------------------------------------
__global__ __launch_bounds__(128)
void lora_down_kernel(const float* __restrict__ X, const float* __restrict__ down,
                      float* __restrict__ tmp)
{
    const int idx  = blockIdx.x * 4 + (threadIdx.x >> 5);
    const int lane = threadIdx.x & 31;
    const int row  = idx >> 4;
    const int r    = idx & 15;
    const float* x = X    + (size_t)row * DM_;
    const float* d = down + (size_t)r * DM_;
    float s = 0.0f;
    for (int k = lane * 4; k < DM_; k += 128) {
        float4 a = *(const float4*)(x + k);
        float4 b = *(const float4*)(d + k);
        s += a.x * b.x + a.y * b.y + a.z * b.z + a.w * b.w;
    }
    #pragma unroll
    for (int m = 1; m < 32; m <<= 1) s += __shfl_xor(s, m);
    if (lane == 0) tmp[idx] = s;
}

// ---------------------------------------------------------------------------
// LoRA up:  X[row][c] = beta*X[row][c] + scale * sum_r tmp[row][r]*up[c][r]
// up is (3072 x 16) row-major.  Thread per element.
// ---------------------------------------------------------------------------
__global__ void lora_up_kernel(const float* __restrict__ tmp, const float* __restrict__ up,
                               float* __restrict__ X, int total, float scale, int beta)
{
    const int idx = blockIdx.x * 256 + threadIdx.x;
    if (idx >= total) return;
    const int row = idx / DM_;
    const int c   = idx % DM_;
    const float* t = tmp + (size_t)row * R_;
    const float* u = up  + (size_t)c * R_;
    float s = 0.0f;
    #pragma unroll
    for (int r = 0; r < R_; ++r) s += t[r] * u[r];
    X[idx] = (beta ? X[idx] : 0.0f) + scale * s;
}

// ---------------------------------------------------------------------------
// RMS-norm + RoPE + concat-pack into [H][S_ALL][DH] layout.
// One wave per (h, s) row; lane handles 4 contiguous channels.
// w_enc/w_img == nullptr => no norm (V path).  doRope=0 => no rotation.
// outScale folds the 1/sqrt(DH) softmax scale into Q.
// ---------------------------------------------------------------------------
__global__ __launch_bounds__(128)
void pack_qkv_kernel(const float* __restrict__ enc_src, const float* __restrict__ img_src,
                     const float* __restrict__ w_enc, const float* __restrict__ w_img,
                     const float* __restrict__ rope_cos, const float* __restrict__ rope_sin,
                     float* __restrict__ dst, int doRope, float outScale)
{
    const int row  = blockIdx.x * 4 + (threadIdx.x >> 5);   // h*S_ALL + s
    const int lane = threadIdx.x & 31;
    const int h = row / S_ALL;
    const int s = row % S_ALL;

    const float* src;
    const float* w;
    if (s < S_TXT) { src = enc_src + (size_t)s * DM_ + h * DH_;           w = w_enc; }
    else           { src = img_src + (size_t)(s - S_TXT) * DM_ + h * DH_; w = w_img; }

    float4 x = *(const float4*)(src + lane * 4);
    if (w) {
        float ss = x.x * x.x + x.y * x.y + x.z * x.z + x.w * x.w;
        #pragma unroll
        for (int m = 1; m < 32; m <<= 1) ss += __shfl_xor(ss, m);
        const float rn = rsqrtf(ss * (1.0f / (float)DH_) + 1e-6f);
        const float4 wv = *(const float4*)(w + lane * 4);
        x.x *= rn * wv.x; x.y *= rn * wv.y; x.z *= rn * wv.z; x.w *= rn * wv.w;
    }
    if (doRope) {
        const float4 c  = *(const float4*)(rope_cos + (size_t)s * DH_ + lane * 4);
        const float4 sn = *(const float4*)(rope_sin + (size_t)s * DH_ + lane * 4);
        float4 o;
        o.x = x.x * c.x - x.y * sn.x;
        o.y = x.y * c.y + x.x * sn.y;
        o.z = x.z * c.z - x.w * sn.z;
        o.w = x.w * c.w + x.z * sn.w;
        x = o;
    }
    x.x *= outScale; x.y *= outScale; x.z *= outScale; x.w *= outScale;
    *(float4*)(dst + ((size_t)h * S_ALL + s) * DH_ + lane * 4) = x;
}

// ---------------------------------------------------------------------------
// Flash attention.  Block = 4 waves on the SAME head, consecutive 16-query
// tiles.  Per 64-key iteration the block cooperatively stages the K tile
// (64x128, [key][dh]) and the transposed V tile (128x64, [dh][key]) into LDS
// as bf16, then each wave builds WMMA fragments with paired ds_load_b32 reads.
// Q is pre-scaled by 1/sqrt(DH).  Output written as [s][h*DH + d].
// ---------------------------------------------------------------------------
#define LDK 130   // K-tile row stride (bf16): odd dword stride -> conflict-free
#define LDV 66    // V^T-tile row stride (bf16)

__global__ __launch_bounds__(128)
void attn_kernel(const float* __restrict__ Qp, const float* __restrict__ Kp,
                 const float* __restrict__ Vp, float* __restrict__ out)
{
    __shared__ __align__(16) bf16 Ks[64 * LDK];     // 16.6 KB
    __shared__ __align__(16) bf16 Vt[128 * LDV];    // 16.9 KB
    __shared__ __align__(16) bf16 Pl[4][16 * 66];   // per-wave P tile

    const int tid  = threadIdx.x;
    const int lane = tid & 31;
    const int wave = tid >> 5;
    const int h    = blockIdx.x / 40;                       // 40 blocks per head
    const int qb   = ((blockIdx.x % 40) * 4 + wave) * 16;   // query tile base
    const int ml   = lane & 15;
    const int kg   = lane >> 4;

    const float* Qh = Qp + (size_t)h * S_ALL * DH_;
    const float* Kh = Kp + (size_t)h * S_ALL * DH_;
    const float* Vh = Vp + (size_t)h * S_ALL * DH_;
    bf16* myP = Pl[wave];

    // staging pattern: whole wave reads one contiguous 512B span of one row
    const int grow = tid >> 5;            // 0..3
    const int gcol = (tid & 31) * 4;      // 0..124

    // Persistent Q fragments: 4 dh-chunks of 32.
    Frag qf[4];
    #pragma unroll
    for (int c = 0; c < 4; ++c) {
        const float* qrow = Qh + (size_t)(qb + ml) * DH_ + c * 32;
        #pragma unroll
        for (int j = 0; j < 8; ++j) {
            const int kk = ((j < 4) ? 2 * j : 16 + 2 * (j - 4)) + kg * 8;
            float2 f = *(const float2*)(qrow + kk);
            qf[c].e[2 * j]     = (bf16)f.x;
            qf[c].e[2 * j + 1] = (bf16)f.y;
        }
    }

    floatx8 Oa[8];
    #pragma unroll
    for (int t = 0; t < 8; ++t)
        #pragma unroll
        for (int r = 0; r < 8; ++r) Oa[t][r] = 0.0f;
    float mrow[8], lrow[8];
    #pragma unroll
    for (int r = 0; r < 8; ++r) { mrow[r] = -1e30f; lrow[r] = 0.0f; }

    for (int kb = 0; kb < S_ALL; kb += 64) {
        // ---- cooperative staging of K (as-is) and V (transposed) ----
        __syncthreads();                  // previous iteration's reads done
        #pragma unroll
        for (int i = 0; i < 16; ++i) {
            const int row = grow + 4 * i;                       // 0..63
            float4 fk = *(const float4*)(Kh + (size_t)(kb + row) * DH_ + gcol);
            float4 fv = *(const float4*)(Vh + (size_t)(kb + row) * DH_ + gcol);
            bf16* dk = Ks + row * LDK + gcol;
            dk[0] = (bf16)fk.x; dk[1] = (bf16)fk.y; dk[2] = (bf16)fk.z; dk[3] = (bf16)fk.w;
            Vt[(gcol + 0) * LDV + row] = (bf16)fv.x;
            Vt[(gcol + 1) * LDV + row] = (bf16)fv.y;
            Vt[(gcol + 2) * LDV + row] = (bf16)fv.z;
            Vt[(gcol + 3) * LDV + row] = (bf16)fv.w;
        }
        __syncthreads();

        // prefetch next key block while we compute on this one
        if (kb + 64 < S_ALL) {
            __builtin_prefetch(Kh + (size_t)(kb + 64 + grow) * DH_ + gcol, 0, 1);
            __builtin_prefetch(Vh + (size_t)(kb + 64 + grow) * DH_ + gcol, 0, 1);
        }

        // ---- scores: S[16 x 64] = Q @ K^T (4 key sub-tiles of 16) ----
        floatx8 Sa[4];
        #pragma unroll
        for (int nc = 0; nc < 4; ++nc) {
            #pragma unroll
            for (int r = 0; r < 8; ++r) Sa[nc][r] = 0.0f;
            #pragma unroll
            for (int c = 0; c < 4; ++c) {
                Frag bfg;
                const bf16* krow = Ks + (nc * 16 + ml) * LDK + c * 32;
                #pragma unroll
                for (int j = 0; j < 8; ++j)
                    bfg.u[j] = *(const unsigned int*)(krow + ((j < 4) ? 2 * j : 16 + 2 * (j - 4)) + kg * 8);
                Sa[nc] = WMMA_BF16(qf[c].v, bfg.v, Sa[nc]);
            }
        }

        // ---- online softmax (row stats via half-wave shfl reductions) ----
        float mnew[8], alpha[8];
        #pragma unroll
        for (int r = 0; r < 8; ++r) {
            float v = fmaxf(fmaxf(Sa[0][r], Sa[1][r]), fmaxf(Sa[2][r], Sa[3][r]));
            #pragma unroll
            for (int d = 1; d < 16; d <<= 1) v = fmaxf(v, __shfl_xor(v, d));
            mnew[r]  = fmaxf(mrow[r], v);
            alpha[r] = __expf(mrow[r] - mnew[r]);
            mrow[r]  = mnew[r];
        }
        #pragma unroll
        for (int r = 0; r < 8; ++r) {
            float s = 0.0f;
            #pragma unroll
            for (int nc = 0; nc < 4; ++nc) {
                const float p = __expf(Sa[nc][r] - mnew[r]);
                Sa[nc][r] = p;
                s += p;
            }
            #pragma unroll
            for (int d = 1; d < 16; d <<= 1) s += __shfl_xor(s, d);
            lrow[r] = lrow[r] * alpha[r] + s;
        }

        // ---- P -> LDS (bf16, C-layout scatter), rescale O ----
        #pragma unroll
        for (int nc = 0; nc < 4; ++nc)
            #pragma unroll
            for (int r = 0; r < 8; ++r)
                myP[(r + 8 * kg) * 66 + nc * 16 + ml] = (bf16)Sa[nc][r];
        asm volatile("s_wait_dscnt 0" ::: "memory");

        #pragma unroll
        for (int t = 0; t < 8; ++t)
            #pragma unroll
            for (int r = 0; r < 8; ++r) Oa[t][r] *= alpha[r];

        // ---- O += P @ V  (2 key chunks of 32, 8 dh tiles of 16) ----
        #pragma unroll
        for (int kc = 0; kc < 2; ++kc) {
            Frag pf;
            #pragma unroll
            for (int j = 0; j < 8; ++j) {
                const int kk = ((j < 4) ? 2 * j : 16 + 2 * (j - 4)) + kg * 8 + kc * 32;
                pf.u[j] = *(const unsigned int*)(myP + ml * 66 + kk);
            }
            #pragma unroll
            for (int t = 0; t < 8; ++t) {
                Frag vf;
                const bf16* vrow = Vt + (t * 16 + ml) * LDV + kc * 32;
                #pragma unroll
                for (int j = 0; j < 8; ++j)
                    vf.u[j] = *(const unsigned int*)(vrow + ((j < 4) ? 2 * j : 16 + 2 * (j - 4)) + kg * 8);
                Oa[t] = WMMA_BF16(pf.v, vf.v, Oa[t]);
            }
        }
    }

    // ---- normalize and write out[s][h*DH + d] ----
    #pragma unroll
    for (int t = 0; t < 8; ++t)
        #pragma unroll
        for (int r = 0; r < 8; ++r) {
            const int s  = qb + r + 8 * kg;
            const int dh = t * 16 + ml;
            out[(size_t)s * DM_ + h * DH_ + dh] = Oa[t][r] / lrow[r];
        }
}

// ---------------------------------------------------------------------------
// Bokeh cross-attention: 4 keys only -> plain VALU, adds into attention out.
// One wave per (h, s) row.  Q already carries the 1/sqrt(DH) scale.
// ---------------------------------------------------------------------------
__global__ __launch_bounds__(128)
void bokeh_attn_kernel(const float* __restrict__ Qp, const float* __restrict__ kd,
                       const float* __restrict__ vd, float* __restrict__ out)
{
    const int row  = blockIdx.x * 4 + (threadIdx.x >> 5);
    const int lane = threadIdx.x & 31;
    const int h = row / S_ALL;
    const int s = row % S_ALL;

    const float4 qv = *(const float4*)(Qp + ((size_t)h * S_ALL + s) * DH_ + lane * 4);
    float sc[4];
    #pragma unroll
    for (int j = 0; j < 4; ++j) {
        const float4 kv = *(const float4*)(kd + (size_t)j * DM_ + h * DH_ + lane * 4);
        float d = qv.x * kv.x + qv.y * kv.y + qv.z * kv.z + qv.w * kv.w;
        #pragma unroll
        for (int m = 1; m < 32; m <<= 1) d += __shfl_xor(d, m);
        sc[j] = d;
    }
    const float mx = fmaxf(fmaxf(sc[0], sc[1]), fmaxf(sc[2], sc[3]));
    float e[4], sum = 0.0f;
    #pragma unroll
    for (int j = 0; j < 4; ++j) { e[j] = __expf(sc[j] - mx); sum += e[j]; }
    const float inv = 1.0f / sum;

    float4 o = {0.0f, 0.0f, 0.0f, 0.0f};
    #pragma unroll
    for (int j = 0; j < 4; ++j) {
        const float4 vv = *(const float4*)(vd + (size_t)j * DM_ + h * DH_ + lane * 4);
        const float w = e[j] * inv;
        o.x += w * vv.x; o.y += w * vv.y; o.z += w * vv.z; o.w += w * vv.w;
    }
    float* op = out + (size_t)s * DM_ + h * DH_ + lane * 4;
    float4 cur = *(const float4*)op;
    cur.x += o.x; cur.y += o.y; cur.z += o.z; cur.w += o.w;   // BOKEH_SCALE = 1
    *(float4*)op = cur;
}

// ---------------------------------------------------------------------------
// Host-side launcher
// ---------------------------------------------------------------------------
extern "C" void kernel_launch(void* const* d_in, const int* in_sizes, int n_in,
                              void* d_out, int out_size, void* d_ws, size_t ws_size,
                              hipStream_t stream)
{
    (void)in_sizes; (void)n_in; (void)out_size; (void)ws_size;

    const float* hs   = (const float*)d_in[0];   // (2048, 3072)
    const float* enc  = (const float*)d_in[1];   // (512, 3072)
    const float* bok  = (const float*)d_in[2];   // (4, 3072)
    const float* rc   = (const float*)d_in[3];   // (2560, 128)
    const float* rs   = (const float*)d_in[4];   // (2560, 128)
    const float* Wq   = (const float*)d_in[5];
    const float* Wk   = (const float*)d_in[6];
    const float* Wv   = (const float*)d_in[7];
    const float* Waq  = (const float*)d_in[8];
    const float* Wak  = (const float*)d_in[9];
    const float* Wav  = (const float*)d_in[10];
    const float* Wo   = (const float*)d_in[11];
    const float* Wao  = (const float*)d_in[12];
    const float* bq   = (const float*)d_in[13];
    const float* bk   = (const float*)d_in[14];
    const float* bv   = (const float*)d_in[15];
    const float* baq  = (const float*)d_in[16];
    const float* bak  = (const float*)d_in[17];
    const float* bav  = (const float*)d_in[18];
    const float* bo   = (const float*)d_in[19];
    const float* bao  = (const float*)d_in[20];
    const float* nqw  = (const float*)d_in[21];
    const float* nkw  = (const float*)d_in[22];
    const float* naqw = (const float*)d_in[23];
    const float* nakw = (const float*)d_in[24];
    const float* kb_down   = (const float*)d_in[25];
    const float* vb_down   = (const float*)d_in[26];
    const float* qadp_down = (const float*)d_in[27];
    const float* kadp_down = (const float*)d_in[28];
    const float* kb_up     = (const float*)d_in[29];
    const float* vb_up     = (const float*)d_in[30];
    const float* qadp_up   = (const float*)d_in[31];
    const float* kadp_up   = (const float*)d_in[32];

    float* outp = (float*)d_out;                 // img_out (2048x3072) ++ enc_out (512x3072)

    // Workspace layout (~210 MB of fp32).
    float* ws = (float*)d_ws;
    size_t off = 0;
    float* q_img = ws + off; off += (size_t)S_IMG * DM_;
    float* k_img = ws + off; off += (size_t)S_IMG * DM_;
    float* v_img = ws + off; off += (size_t)S_IMG * DM_;
    float* eq    = ws + off; off += (size_t)S_TXT * DM_;
    float* ek    = ws + off; off += (size_t)S_TXT * DM_;
    float* ev    = ws + off; off += (size_t)S_TXT * DM_;
    float* Qp    = ws + off; off += (size_t)H_ * S_ALL * DH_;
    float* Kp    = ws + off; off += (size_t)H_ * S_ALL * DH_;
    float* Vp    = ws + off; off += (size_t)H_ * S_ALL * DH_;
    float* attn  = ws + off; off += (size_t)S_ALL * DM_;
    float* ltmp  = ws + off; off += (size_t)S_IMG * R_;
    float* btk   = ws + off; off += 4 * R_;
    float* btv   = ws + off; off += 4 * R_;
    float* kdb   = ws + off; off += (size_t)4 * DM_;
    float* vdb   = ws + off; off += (size_t)4 * DM_;

    const dim3 gemmBlk(128);
    const dim3 gImg(DM_ / 128, S_IMG / 128);     // (24, 16)
    const dim3 gEnc(DM_ / 128, S_TXT / 128);     // (24, 4)
    const float scl = 1.0f / sqrtf((float)DH_);

    // 1. image-stream QKV projections (bf16 WMMA)
    gemm_bias_kernel<<<gImg, gemmBlk, 0, stream>>>(hs, Wq, bq, q_img, S_IMG, DM_, DM_);
    gemm_bias_kernel<<<gImg, gemmBlk, 0, stream>>>(hs, Wk, bk, k_img, S_IMG, DM_, DM_);
    gemm_bias_kernel<<<gImg, gemmBlk, 0, stream>>>(hs, Wv, bv, v_img, S_IMG, DM_, DM_);

    // 2. LoRA adapters on q and k (input is post-projection q/k)
    lora_down_kernel<<<S_IMG * R_ / 4, 128, 0, stream>>>(q_img, qadp_down, ltmp);
    lora_up_kernel<<<(S_IMG * DM_ + 255) / 256, 256, 0, stream>>>(ltmp, qadp_up, q_img,
                                                                  S_IMG * DM_, 1.0f, 1);
    lora_down_kernel<<<S_IMG * R_ / 4, 128, 0, stream>>>(k_img, kadp_down, ltmp);
    lora_up_kernel<<<(S_IMG * DM_ + 255) / 256, 256, 0, stream>>>(ltmp, kadp_up, k_img,
                                                                  S_IMG * DM_, 1.0f, 1);

    // 3. encoder-stream QKV projections
    gemm_bias_kernel<<<gEnc, gemmBlk, 0, stream>>>(enc, Waq, baq, eq, S_TXT, DM_, DM_);
    gemm_bias_kernel<<<gEnc, gemmBlk, 0, stream>>>(enc, Wak, bak, ek, S_TXT, DM_, DM_);
    gemm_bias_kernel<<<gEnc, gemmBlk, 0, stream>>>(enc, Wav, bav, ev, S_TXT, DM_, DM_);

    // 4. rms-norm + RoPE + concat-pack into [H][S][DH]; fold 1/sqrt(DH) into Q
    const int packBlocks = H_ * S_ALL / 4;       // 15360
    pack_qkv_kernel<<<packBlocks, 128, 0, stream>>>(eq, q_img, naqw, nqw, rc, rs, Qp, 1, scl);
    pack_qkv_kernel<<<packBlocks, 128, 0, stream>>>(ek, k_img, nakw, nkw, rc, rs, Kp, 1, 1.0f);
    pack_qkv_kernel<<<packBlocks, 128, 0, stream>>>(ev, v_img, nullptr, nullptr, rc, rs, Vp, 0, 1.0f);

    // 5. main SDPA: 24 heads x 160 query tiles, 4 waves/block -> 960 blocks
    attn_kernel<<<H_ * 40, 128, 0, stream>>>(Qp, Kp, Vp, attn);

    // 6. bokeh branch: LoRA-projected 4-token K/V, tiny attention, add into attn
    lora_down_kernel<<<4 * R_ / 4, 128, 0, stream>>>(bok, kb_down, btk);
    lora_down_kernel<<<4 * R_ / 4, 128, 0, stream>>>(bok, vb_down, btv);
    lora_up_kernel<<<(4 * DM_ + 255) / 256, 256, 0, stream>>>(btk, kb_up, kdb, 4 * DM_, 1.0f, 0);
    lora_up_kernel<<<(4 * DM_ + 255) / 256, 256, 0, stream>>>(btv, vb_up, vdb, 4 * DM_, 1.0f, 0);
    bokeh_attn_kernel<<<H_ * S_ALL / 4, 128, 0, stream>>>(Qp, kdb, vdb, attn);

    // 7. output projections straight into d_out (img first, then enc)
    gemm_bias_kernel<<<gImg, gemmBlk, 0, stream>>>(attn + (size_t)S_TXT * DM_, Wo, bo,
                                                   outp, S_IMG, DM_, DM_);
    gemm_bias_kernel<<<gEnc, gemmBlk, 0, stream>>>(attn, Wao, bao,
                                                   outp + (size_t)S_IMG * DM_, S_TXT, DM_, DM_);
}
